// MultiHeadAttention_85899346206
// MI455X (gfx1250) — matrix operand order repaired and checked by reading, not compiled, not run on hardware
//
#include <hip/hip_runtime.h>
#include <hip/hip_bf16.h>
#include <stdint.h>

typedef __attribute__((ext_vector_type(16))) __bf16 v16bf;
typedef __attribute__((ext_vector_type(8)))  __bf16 v8bf;
typedef __attribute__((ext_vector_type(8)))  float  v8f;

#define S_LEN 4096
#define DM    512
#define HEADS 8
#define DK    64

// ---- WMMA helper: D = A(16x32 bf16) x B(32x16 bf16) + C(16x16 f32) ----
static __device__ __forceinline__ v8f mma_bf16(v16bf a, v16bf b, v8f c) {
  return __builtin_amdgcn_wmma_f32_16x16x32_bf16(
      /*neg_a=*/false, a, /*neg_b=*/false, b,
      /*c_mod=*/(short)0, c, /*reuse_a=*/false, /*reuse_b=*/false);
}

static __device__ __forceinline__ v16bf cat8(v8bf lo, v8bf hi) {
  return __builtin_shufflevector(lo, hi, 0,1,2,3,4,5,6,7,8,9,10,11,12,13,14,15);
}

// A-fragment (16x32 bf16, row-major source). Lane L: row M=L%16; per lane two
// contiguous 16B runs at K = 8*(L/16) and K = 16 + 8*(L/16).
static __device__ __forceinline__ v16bf load_a_bf16(const __bf16* base, int stride) {
  const int lane = threadIdx.x & 31;
  const int row  = lane & 15, hi = lane >> 4;
  const __bf16* p = base + row * stride + hi * 8;
  v8bf x = *(const v8bf*)(p);
  v8bf y = *(const v8bf*)(p + 16);
  return cat8(x, y);
}

// B-fragment (32x16 bf16) from B^T stored row-major [N][K]. Lane L: col N=L%16;
// two contiguous 16B runs at K = 8*(L/16) and K = 16 + 8*(L/16).
static __device__ __forceinline__ v16bf load_bt_bf16(const __bf16* base, int stride) {
  const int lane = threadIdx.x & 31;
  const int n = lane & 15, hi = lane >> 4;
  const __bf16* p = base + n * stride + hi * 8;
  v8bf x = *(const v8bf*)(p);
  v8bf y = *(const v8bf*)(p + 16);
  return cat8(x, y);
}

// fp32-source A-fragment (convert in flight; lowers to v_cvt_pk_bf16_f32)
static __device__ __forceinline__ v16bf load_a_f32(const float* base, int stride) {
  const int lane = threadIdx.x & 31;
  const int row  = lane & 15, hi = lane >> 4;
  const float* p = base + row * stride + hi * 8;
  v16bf r;
#pragma unroll
  for (int i = 0; i < 8; ++i) r[i] = (__bf16)p[i];
#pragma unroll
  for (int i = 0; i < 8; ++i) r[i + 8] = (__bf16)p[i + 16];
  return r;
}

// ---- CDNA5 async global->LDS copy (ASYNCcnt-tracked), 16B per lane ----
static __device__ __forceinline__ void async_ld_b128(uint32_t ldsOff, const void* gp) {
  asm volatile("global_load_async_to_lds_b128 %0, %1, off"
               :: "v"(ldsOff), "v"(gp) : "memory");
}
static __device__ __forceinline__ void wait_async0() {
  asm volatile("s_wait_asynccnt 0x0" ::: "memory");
}
static __device__ __forceinline__ uint32_t lds_off(const void* p) {
  return (uint32_t)(uintptr_t)p;  // flat->LDS is a truncate (aperture mapping)
}

// =====================================================================
// Kernel 0: one-shot weight conversion fp32 -> bf16 (Wq|Wk|Wv|Wo).
// grid = (DM*DM/1024, 4), block = 256; each thread converts 4 elements.
// =====================================================================
__global__ __launch_bounds__(256) void cvt_w_kernel(
    const float* __restrict__ Wq, const float* __restrict__ Wk,
    const float* __restrict__ Wv, const float* __restrict__ Wo,
    __bf16* __restrict__ dst) {
  const float* src = (blockIdx.y == 0) ? Wq : (blockIdx.y == 1) ? Wk
                   : (blockIdx.y == 2) ? Wv : Wo;
  const size_t idx = ((size_t)blockIdx.x * 256 + threadIdx.x) * 4;
  const float4 vv  = *(const float4*)(src + idx);
  __bf16* o = dst + (size_t)blockIdx.y * DM * DM + idx;
  o[0] = (__bf16)vv.x; o[1] = (__bf16)vv.y;
  o[2] = (__bf16)vv.z; o[3] = (__bf16)vv.w;
}

// =====================================================================
// Kernel 1: fused QKV projection. Y = X @ W^T + b, split heads, bf16 out.
// grid = (M/128, N/64, 3), block = 128 (4 waves); wave computes 32x64
// (2 strips of 16 rows -> each bf16 B-fragment feeds two WMMAs).
// Q,K stored [bh][s][dk] row-major; V stored transposed [bh][dk][s].
// =====================================================================
__global__ __launch_bounds__(128) void qkv_proj_kernel(
    const float* __restrict__ q, const float* __restrict__ k,
    const float* __restrict__ v, const __bf16* __restrict__ Wb,
    const float* __restrict__ bq, const float* __restrict__ bk,
    const float* __restrict__ bv,
    __bf16* __restrict__ Qb, __bf16* __restrict__ Kb, __bf16* __restrict__ Vt) {
  const int wave  = threadIdx.x >> 5;
  const int lane  = threadIdx.x & 31;
  const int mBase = blockIdx.x * 128 + wave * 32;
  const int nBase = blockIdx.y * 64;
  const int which = blockIdx.z;

  const float*  X    = (which == 0) ? q  : (which == 1) ? k  : v;
  const __bf16* W    = Wb + (size_t)which * DM * DM;  // bf16 weights
  const float*  bias = (which == 0) ? bq : (which == 1) ? bk : bv;

  v8f acc[2][4] = {{v8f{}, v8f{}, v8f{}, v8f{}}, {v8f{}, v8f{}, v8f{}, v8f{}}};
  for (int kc = 0; kc < DM; kc += 32) {
    v16bf aF0 = load_a_f32(X + (size_t)(mBase +  0) * DM + kc, DM);
    v16bf aF1 = load_a_f32(X + (size_t)(mBase + 16) * DM + kc, DM);
#pragma unroll
    for (int n = 0; n < 4; ++n) {
      v16bf bF = load_bt_bf16(W + (size_t)(nBase + n * 16) * DM + kc, DM);
      acc[0][n] = mma_bf16(aF0, bF, acc[0][n]);
      acc[1][n] = mma_bf16(aF1, bF, acc[1][n]);
    }
  }

  const int cn = lane & 15, hi = lane >> 4;
#pragma unroll
  for (int n = 0; n < 4; ++n) {
    const int   c  = nBase + n * 16 + cn;
    const float bb = bias[c];
    const int   h  = c >> 6, d = c & 63;
#pragma unroll
    for (int s = 0; s < 2; ++s)
#pragma unroll
      for (int r = 0; r < 8; ++r) {
        const int   m   = mBase + 16 * s + r + 8 * hi;  // row in [B*S]
        const int   bt  = m >> 12, sq = m & 4095;       // batch, seq
        const int   bh  = bt * HEADS + h;
        const float val = acc[s][n][r] + bb;
        if (which == 0)
          Qb[((size_t)bh * S_LEN + sq) * DK + d] = (__bf16)val;
        else if (which == 1)
          Kb[((size_t)bh * S_LEN + sq) * DK + d] = (__bf16)val;
        else
          Vt[((size_t)bh * DK + d) * S_LEN + sq] = (__bf16)val;
      }
  }
}

// =====================================================================
// Kernel 2: flash attention. grid = (S/128, B*H), block = 128 (4 waves).
// Wave owns 32 query rows (2 strips of 16). K/V tiles (32 keys) staged into
// double-buffered LDS with async global->LDS copies, shared by all waves;
// prefetch of tile i+1 overlaps the 16 WMMAs/wave of tile i.
// =====================================================================
#define NIT (S_LEN / 32)

__global__ __launch_bounds__(128) void flash_attn_kernel(
    const __bf16* __restrict__ Qb, const __bf16* __restrict__ Kb,
    const __bf16* __restrict__ Vt, __bf16* __restrict__ Ctx) {
  __shared__ __align__(16) __bf16 kbuf[2][32 * 64];  // [t][d], stride 64
  __shared__ __align__(16) __bf16 vbuf[2][64 * 32];  // [d][t], stride 32
  __shared__ __align__(16) __bf16 pbuf[4][16 * 32];  // per-wave P staging

  const int tid  = threadIdx.x;
  const int wave = tid >> 5;
  const int lane = tid & 31;
  const int cn = lane & 15, hi = lane >> 4;
  const int bh   = blockIdx.y;
  const int qRow = blockIdx.x * 128 + wave * 32;

  const __bf16* Qh = Qb + (size_t)bh * S_LEN * DK;
  const __bf16* Kh = Kb + (size_t)bh * S_LEN * DK;
  const __bf16* Vh = Vt + (size_t)bh * DK * S_LEN;
  __bf16* myP = &pbuf[wave][0];

  // Q fragments for both strips stay in registers (32 rows x 64 d).
  v16bf qA[2][2];
#pragma unroll
  for (int s = 0; s < 2; ++s)
#pragma unroll
    for (int c = 0; c < 2; ++c)
      qA[s][c] = load_a_bf16(Qh + (size_t)(qRow + 16 * s) * DK + 32 * c, DK);

  float mrow[2][8], lrow[2][8];
#pragma unroll
  for (int s = 0; s < 2; ++s)
#pragma unroll
    for (int r = 0; r < 8; ++r) { mrow[s][r] = -3.0e38f; lrow[s][r] = 0.0f; }
  v8f ctx[2][4] = {{v8f{}, v8f{}, v8f{}, v8f{}}, {v8f{}, v8f{}, v8f{}, v8f{}}};

  // Cooperative async stage of one 32-key tile: K 4KB + V^T 4KB.
  auto stage = [&](int buf, int t0) {
#pragma unroll
    for (int j = 0; j < 2; ++j) {
      const int ck = tid * 2 + j;                // 0..255
      const int krow = ck >> 3, kpart = ck & 7;  // K tile rows are 128B
      async_ld_b128(lds_off(&kbuf[buf][krow * 64 + kpart * 8]),
                    Kh + (size_t)(t0 + krow) * DK + kpart * 8);
      const int vrow = ck >> 2, vpart = ck & 3;  // V tile rows are 64B
      async_ld_b128(lds_off(&vbuf[buf][vrow * 32 + vpart * 8]),
                    Vh + (size_t)vrow * S_LEN + t0 + vpart * 8);
    }
  };
  stage(0, 0);

  const float cexp = 0.125f * 1.44269504088896f;  // (1/sqrt(64)) * log2(e)

  for (int it = 0; it < NIT; ++it) {
    const int t0  = it * 32;
    const int cur = it & 1;
    wait_async0();      // our slice of tile `it` is in LDS
    __syncthreads();    // everyone's slice is in LDS; prior reads drained
    if (it + 1 < NIT) stage(cur ^ 1, t0 + 32);  // overlap next fetch

    const __bf16* kt = &kbuf[cur][0];
    const __bf16* vt = &vbuf[cur][0];

    // scores: each K fragment feeds both row strips
    v8f sc[2][2] = {{v8f{}, v8f{}}, {v8f{}, v8f{}}};
#pragma unroll
    for (int tt = 0; tt < 2; ++tt)
#pragma unroll
      for (int c = 0; c < 2; ++c) {
        v16bf kB = load_bt_bf16(kt + (16 * tt) * 64 + 32 * c, 64);
        sc[0][tt] = mma_bf16(qA[0][c], kB, sc[0][tt]);
        sc[1][tt] = mma_bf16(qA[1][c], kB, sc[1][tt]);
      }

    // online softmax per strip + C-frag -> A-frag relayout via private LDS
    v16bf pA[2];
#pragma unroll
    for (int s = 0; s < 2; ++s) {
#pragma unroll
      for (int r = 0; r < 8; ++r) {
        float mx = fmaxf(sc[s][0][r], sc[s][1][r]);
#pragma unroll
        for (int msk = 1; msk < 16; msk <<= 1) mx = fmaxf(mx, __shfl_xor(mx, msk, 32));
        const float mnew = fmaxf(mrow[s][r], mx);
        const float corr = __builtin_amdgcn_exp2f((mrow[s][r] - mnew) * cexp);
        mrow[s][r] = mnew;
        const float p0 = __builtin_amdgcn_exp2f((sc[s][0][r] - mnew) * cexp);
        const float p1 = __builtin_amdgcn_exp2f((sc[s][1][r] - mnew) * cexp);
        sc[s][0][r] = p0; sc[s][1][r] = p1;
        float rs = p0 + p1;
#pragma unroll
        for (int msk = 1; msk < 16; msk <<= 1) rs += __shfl_xor(rs, msk, 32);
        lrow[s][r] = lrow[s][r] * corr + rs;
        ctx[s][0][r] *= corr; ctx[s][1][r] *= corr;
        ctx[s][2][r] *= corr; ctx[s][3][r] *= corr;
      }
#pragma unroll
      for (int r = 0; r < 8; ++r) {
        myP[(r + 8 * hi) * 32 + cn]      = (__bf16)sc[s][0][r];
        myP[(r + 8 * hi) * 32 + cn + 16] = (__bf16)sc[s][1][r];
      }
      asm volatile("s_wait_dscnt 0x0" ::: "memory");  // DS in-order per wave
      pA[s] = load_a_bf16(myP, 32);
    }

    // ctx += P x V; each V fragment feeds both strips
#pragma unroll
    for (int n = 0; n < 4; ++n) {
      v16bf vB = load_bt_bf16(vt + (n * 16) * 32, 32);
      ctx[0][n] = mma_bf16(pA[0], vB, ctx[0][n]);
      ctx[1][n] = mma_bf16(pA[1], vB, ctx[1][n]);
    }
  }

  // normalize and write ctx in [B*S, D_MODEL] bf16 for the output GEMM
  const int b = bh >> 3, h = bh & 7;
#pragma unroll
  for (int s = 0; s < 2; ++s)
#pragma unroll
    for (int r = 0; r < 8; ++r) {
      const float inv = 1.0f / lrow[s][r];
      const int   sq  = qRow + 16 * s + r + 8 * hi;
#pragma unroll
      for (int n = 0; n < 4; ++n) {
        const float val = ctx[s][n][r] * inv;
        Ctx[((size_t)b * S_LEN + sq) * DM + h * DK + n * 16 + cn] = (__bf16)val;
      }
    }
}

// =====================================================================
// Kernel 3: output projection. out = Ctx @ W_o^T + b_o (fp32 out).
// grid = (M/128, N/64), block = 128; wave computes 32x64.
// =====================================================================
__global__ __launch_bounds__(128) void out_proj_kernel(
    const __bf16* __restrict__ Ctx, const __bf16* __restrict__ Wob,
    const float* __restrict__ bo, float* __restrict__ out) {
  const int wave  = threadIdx.x >> 5;
  const int lane  = threadIdx.x & 31;
  const int mBase = blockIdx.x * 128 + wave * 32;
  const int nBase = blockIdx.y * 64;

  v8f acc[2][4] = {{v8f{}, v8f{}, v8f{}, v8f{}}, {v8f{}, v8f{}, v8f{}, v8f{}}};
  for (int kc = 0; kc < DM; kc += 32) {
    v16bf aF0 = load_a_bf16(Ctx + (size_t)(mBase +  0) * DM + kc, DM);
    v16bf aF1 = load_a_bf16(Ctx + (size_t)(mBase + 16) * DM + kc, DM);
#pragma unroll
    for (int n = 0; n < 4; ++n) {
      v16bf bF = load_bt_bf16(Wob + (size_t)(nBase + n * 16) * DM + kc, DM);
      acc[0][n] = mma_bf16(aF0, bF, acc[0][n]);
      acc[1][n] = mma_bf16(aF1, bF, acc[1][n]);
    }
  }

  const int cn = lane & 15, hi = lane >> 4;
#pragma unroll
  for (int n = 0; n < 4; ++n) {
    const int   c  = nBase + n * 16 + cn;
    const float bb = bo[c];
#pragma unroll
    for (int s = 0; s < 2; ++s)
#pragma unroll
      for (int r = 0; r < 8; ++r) {
        const int m = mBase + 16 * s + r + 8 * hi;
        out[(size_t)m * DM + c] = acc[s][n][r] + bb;
      }
  }
}

// =====================================================================
extern "C" void kernel_launch(void* const* d_in, const int* in_sizes, int n_in,
                              void* d_out, int out_size, void* d_ws, size_t ws_size,
                              hipStream_t stream) {
  const float* q  = (const float*)d_in[0];
  const float* k  = (const float*)d_in[1];
  const float* v  = (const float*)d_in[2];
  const float* Wq = (const float*)d_in[3];
  const float* bq = (const float*)d_in[4];
  const float* Wk = (const float*)d_in[5];
  const float* bk = (const float*)d_in[6];
  const float* Wv = (const float*)d_in[7];
  const float* bv = (const float*)d_in[8];
  const float* Wo = (const float*)d_in[9];
  const float* bo = (const float*)d_in[10];
  float* out = (float*)d_out;

  // workspace: Qb | Kb | Vt | Ctx (8 MB each) | Wb (bf16 Wq,Wk,Wv,Wo; 2 MB)
  const size_t SEG = (size_t)2 * HEADS * S_LEN * DK * sizeof(__bf16);
  __bf16* Qb = (__bf16*)((char*)d_ws + 0 * SEG);
  __bf16* Kb = (__bf16*)((char*)d_ws + 1 * SEG);
  __bf16* Vt = (__bf16*)((char*)d_ws + 2 * SEG);
  __bf16* Cb = (__bf16*)((char*)d_ws + 3 * SEG);
  __bf16* Wb = (__bf16*)((char*)d_ws + 4 * SEG);

  dim3 blk128(128);
  cvt_w_kernel<<<dim3(DM * DM / 1024, 4), dim3(256), 0, stream>>>(Wq, Wk, Wv, Wo, Wb);
  qkv_proj_kernel<<<dim3(64, 8, 3), blk128, 0, stream>>>(q, k, v, Wb, bq, bk, bv,
                                                         Qb, Kb, Vt);
  flash_attn_kernel<<<dim3(32, 16), blk128, 0, stream>>>(Qb, Kb, Vt, Cb);
  out_proj_kernel<<<dim3(64, 8), blk128, 0, stream>>>(Cb, Wb + (size_t)3 * DM * DM,
                                                      bo, out);
}
